// Diffuser_7335804142350
// MI455X (gfx1250) — compile-verified, matrix operand order checked by reading, not run on hardware
//
#include <hip/hip_runtime.h>
#include <hip/hip_bf16.h>
#include <math.h>

// ---------------- model dims (match reference) ----------------
#define H      256
#define HEADS  8
#define DH     32
#define A_N    1024
#define S_N    8
#define L_N    10
#define SD_N   4
#define HIST_N 21
#define M_N    8192
#define E1_N   131072
#define ET_N   65536
#define NF_N   (A_N * S_N)     // 8192 future tokens
#define NP_N   (A_N * HIST_N)  // 21504 past tokens

typedef __bf16 bf16;
typedef __attribute__((ext_vector_type(16))) __bf16 v16bf;
typedef __attribute__((ext_vector_type(8)))  float  v8f;
typedef __attribute__((ext_vector_type(4)))  float  v4f;

// ---------------- helpers ----------------
__device__ __forceinline__ unsigned ordf(float x) {
    unsigned u = __float_as_uint(x);
    return (u & 0x80000000u) ? ~u : (u | 0x80000000u);
}
__device__ __forceinline__ float unordf(unsigned u) {
    return (u & 0x80000000u) ? __uint_as_float(u ^ 0x80000000u) : __uint_as_float(~u);
}

// CDNA5 async copy: global -> LDS, tracked by ASYNCcnt (CDNA5 ISA).
// GVS mode: mem_addr = SADDR + VADDR(u32); vdst VGPR holds LDS byte address.
__device__ __forceinline__ void async_g2l_b128(unsigned lds_addr, unsigned voff,
                                               const void* sbase) {
    asm volatile("global_load_async_to_lds_b128 %0, %1, %2"
                 :: "v"(lds_addr), "v"(voff), "s"(sbase)
                 : "memory");
}
__device__ __forceinline__ void wait_async0() {
    asm volatile("s_wait_asynccnt 0x0" ::: "memory");
}

// =====================================================================
// Core WMMA GEMM: out[nrows,256] = act(X[nrows,256] @ Wt^T + bias) (+res)
// X bf16 row-major [nrows][256]; Wt bf16 row-major [256 out][256 k].
// Block = 256 threads = 8 waves; block tile 128(M) x 64(N); each wave does
// a 32x32 tile = 2x2 v_wmma_f32_16x16x32_bf16.
// Tiles staged via GLOBAL_LOAD_ASYNC_TO_LDS_B128 with double-buffered LDS:
// iter kk computes from buf[kk&1] while the async unit fills buf[(kk+1)&1].
// =====================================================================
template<int ACT, bool HAS_BIAS, bool HAS_RES, bool OUT_F32, bool OUT_BF16>
__global__ __launch_bounds__(256)
void gemm256_wmma(const bf16* __restrict__ X, const bf16* __restrict__ Wt,
                  const float* __restrict__ bias, const float* __restrict__ res,
                  float* __restrict__ outF, bf16* __restrict__ outB, int nrows)
{
    __shared__ __align__(16) bf16 lA[2][128 * 40];   // 32 cols + 8 pad
    __shared__ __align__(16) bf16 lB[2][64 * 40];

    const int tid  = threadIdx.x;
    const int lane = tid & 31;
    const int wave = tid >> 5;
    const int wm   = wave & 3;     // M sub-tile group (4)
    const int wn   = wave >> 2;    // N sub-tile group (2)
    const int half = lane >> 4;    // ISA: lanes 16-31 hold K+8 / M+8 halves
    const int r    = lane & 15;
    const int c0   = half * 8;     // bf16 K offset of this lane's first chunk
    const int r0   = blockIdx.x * 128;
    const int n0   = blockIdx.y * 64;

    // ---- per-thread async staging coordinates ----
    // X tile 128x32 bf16: thread covers rows {tid>>2, 64+(tid>>2)}, 16B chunk col
    const int xrow  = tid >> 2;
    const int xcol  = (tid & 3) * 8;                 // bf16 units
    // clamp to a valid row (outputs for padded rows are masked at store)
    const int cr0   = min(r0 + xrow,      nrows - 1) - r0;
    const int cr1   = min(r0 + 64 + xrow, nrows - 1) - r0;
    const unsigned vx0 = (unsigned)cr0 * 512u + (unsigned)xcol * 2u;   // bytes
    const unsigned vx1 = (unsigned)cr1 * 512u + (unsigned)xcol * 2u;
    // W tile 64x32 bf16: one 16B chunk per thread
    const int wrow  = tid >> 2;
    const int wcol  = (tid & 3) * 8;
    const unsigned vw  = (unsigned)wrow * 512u + (unsigned)wcol * 2u;
    const bf16* xbase = X  + (size_t)r0 * 256;
    const bf16* wbase = Wt + (size_t)n0 * 256;

    const unsigned ldsX0 = (unsigned)(uintptr_t)&lA[0][xrow * 40 + xcol];
    const unsigned ldsX1 = (unsigned)(uintptr_t)&lA[0][(64 + xrow) * 40 + xcol];
    const unsigned ldsW  = (unsigned)(uintptr_t)&lB[0][wrow * 40 + wcol];
    const unsigned bufA  = (unsigned)(sizeof(lA) / 2);   // byte stride between A buffers
    const unsigned bufB  = (unsigned)(sizeof(lB) / 2);

    auto prefetch = [&](int kk, int buf) {
        unsigned kb = (unsigned)kk * 64u;                // 32 bf16 = 64B per k-step
        async_g2l_b128(ldsX0 + (unsigned)buf * bufA, vx0 + kb, xbase);
        async_g2l_b128(ldsX1 + (unsigned)buf * bufA, vx1 + kb, xbase);
        async_g2l_b128(ldsW  + (unsigned)buf * bufB, vw  + kb, wbase);
    };

    v8f acc[2][2];
    acc[0][0] = 0; acc[0][1] = 0; acc[1][0] = 0; acc[1][1] = 0;

    prefetch(0, 0);

    for (int kk = 0; kk < 8; ++kk) {
        wait_async0();          // this wave's async tile copies landed in LDS
        __syncthreads();        // every wave's copies landed
        if (kk < 7) prefetch(kk + 1, (kk + 1) & 1);
        const int cur = kk & 1;

        // ---- fragments per ISA 16-bit A 16x32 layout (B identical since
        //      Wt is pre-transposed: rows are output columns) ----
        union Frag { v16bf v; v4f q[2]; };
        Frag a[2], b[2];
        #pragma unroll
        for (int mi = 0; mi < 2; ++mi) {
            int rowA = wm * 32 + mi * 16 + r;
            a[mi].q[0] = *(const v4f*)(&lA[cur][rowA * 40 + c0]);
            a[mi].q[1] = *(const v4f*)(&lA[cur][rowA * 40 + c0 + 16]);
        }
        #pragma unroll
        for (int ni = 0; ni < 2; ++ni) {
            int rowB = wn * 32 + ni * 16 + r;
            b[ni].q[0] = *(const v4f*)(&lB[cur][rowB * 40 + c0]);
            b[ni].q[1] = *(const v4f*)(&lB[cur][rowB * 40 + c0 + 16]);
        }
        #pragma unroll
        for (int mi = 0; mi < 2; ++mi)
            #pragma unroll
            for (int ni = 0; ni < 2; ++ni)
                acc[mi][ni] = __builtin_amdgcn_wmma_f32_16x16x32_bf16(
                    false, a[mi].v, false, b[ni].v, (short)0, acc[mi][ni],
                    false, false);
    }

    // ---- epilogue: C layout = VGPR j -> M=j(+8 for lanes 16-31), N=lane&15
    #pragma unroll
    for (int mi = 0; mi < 2; ++mi) {
        #pragma unroll
        for (int ni = 0; ni < 2; ++ni) {
            int col  = n0 + wn * 32 + ni * 16 + r;
            float bv = HAS_BIAS ? bias[col] : 0.f;
            #pragma unroll
            for (int j = 0; j < 8; ++j) {
                int row = r0 + wm * 32 + mi * 16 + half * 8 + j;
                if (row < nrows) {
                    float v = acc[mi][ni][j] + bv;
                    if (HAS_RES) v += res[(size_t)row * 256 + col];
                    if (ACT == 1) v = fmaxf(v, 0.f);
                    if (ACT == 2) v = v / (1.f + __expf(-v));   // SiLU
                    if (OUT_F32)  outF[(size_t)row * 256 + col] = v;
                    if (OUT_BF16) outB[(size_t)row * 256 + col] = (bf16)v;
                }
            }
        }
    }
}

// =====================================================================
// Misc VALU kernels
// =====================================================================

// Transpose-pack a [256(k)][256(n)] f32 weight into bf16 [n][k].
__global__ void pack_wt_kernel(const float* __restrict__ src, bf16* __restrict__ dst) {
    int n = blockIdx.x, k = threadIdx.x;
    dst[n * 256 + k] = (bf16)src[k * 256 + n];
}

__global__ void f32_to_bf16_kernel(const float* __restrict__ src, bf16* __restrict__ dst,
                                   size_t n) {
    size_t i = (size_t)blockIdx.x * 256 + threadIdx.x;
    if (i < n) dst[i] = (bf16)src[i];
}

__global__ void fill_u32_kernel(unsigned* __restrict__ p, unsigned v, size_t n) {
    size_t i = (size_t)blockIdx.x * 256 + threadIdx.x;
    if (i < n) p[i] = v;
}

// First MLP layer with tiny K: h = relu(feat[row,DIN] @ w1 + b1) -> bf16
template<int DIN>
__global__ __launch_bounds__(256)
void mlp_in_kernel(const float* __restrict__ feat, const float* __restrict__ w1,
                   const float* __restrict__ b1, bf16* __restrict__ out)
{
    __shared__ float sf[64];
    int row = blockIdx.x, o = threadIdx.x;
    if (o < DIN) sf[o] = feat[(size_t)row * DIN + o];
    __syncthreads();
    float acc = b1[o];
    #pragma unroll
    for (int d = 0; d < DIN; ++d) acc = fmaf(sf[d], w1[d * 256 + o], acc);
    out[(size_t)row * 256 + o] = (bf16)fmaxf(acc, 0.f);
}

// Sinusoidal time embedding -> bf16 [A,256]
__global__ void time_sinusoid_kernel(const float* __restrict__ t, bf16* __restrict__ out) {
    int a = blockIdx.x, i = threadIdx.x;
    float tv   = t[a];
    int   j    = i & 127;
    float freq = __expf((float)j * (-9.210340371976184f / 127.f)); // -ln(1e4)/(half-1)
    float ang  = tv * freq;
    out[a * 256 + i] = (bf16)((i < 128) ? __sinf(ang) : __cosf(ang));
}

// future = seg_emb + broadcast(tc_out per agent); dual f32/bf16 output
__global__ void seg_plus_tc_kernel(const float* __restrict__ seg, const float* __restrict__ oA,
                                   float* __restrict__ futF, bf16* __restrict__ futB) {
    size_t idx = (size_t)blockIdx.x * 256 + threadIdx.x;  // NF*256 total
    int tok = (int)(idx >> 8);
    int a   = tok >> 3;                                   // token = agent*S + s
    int col = (int)(idx & 255);
    float v = seg[idx] + oA[(size_t)a * 256 + col];
    futF[idx] = v;
    futB[idx] = (bf16)v;
}

// GAT pass 1: per-(edge,head) score + running segment max (ordered-uint atomicMax)
__global__ void edge_score_kernel(const float* __restrict__ q, const float* __restrict__ k,
                                  const float* __restrict__ ek, const int* __restrict__ src,
                                  const int* __restrict__ dst, float* __restrict__ s,
                                  unsigned* __restrict__ mOrd, int E)
{
    int t = blockIdx.x * 256 + threadIdx.x;
    int e = t >> 3, h = t & 7;
    if (e >= E) return;
    int sN = src[e], dN = dst[e];
    const v4f* qp = (const v4f*)(q  + (size_t)dN * 256 + h * 32);
    const v4f* kp = (const v4f*)(k  + (size_t)sN * 256 + h * 32);
    const v4f* ep = (const v4f*)(ek + (size_t)e  * 256 + h * 32);
    float sum = 0.f;
    #pragma unroll
    for (int i = 0; i < 8; ++i) {
        v4f qv = qp[i], kv = kp[i], ev = ep[i];
        sum += qv.x * (kv.x + ev.x) + qv.y * (kv.y + ev.y)
             + qv.z * (kv.z + ev.z) + qv.w * (kv.w + ev.w);
    }
    sum *= 0.17677669529663687f;  // 1/sqrt(32)
    s[(size_t)e * 8 + h] = sum;
    atomicMax(&mOrd[dN * 8 + h], ordf(sum));
}

// GAT pass 2: alpha = exp(s - max); segment-sum of alpha
__global__ void edge_exp_kernel(const float* __restrict__ s, const int* __restrict__ dst,
                                const unsigned* __restrict__ mOrd, float* __restrict__ al,
                                float* __restrict__ den, int E)
{
    int t = blockIdx.x * 256 + threadIdx.x;
    int e = t >> 3, h = t & 7;
    if (e >= E) return;
    int dN = dst[e];
    float a = __expf(s[(size_t)e * 8 + h] - unordf(mOrd[dN * 8 + h]));
    al[(size_t)e * 8 + h] = a;
    atomicAdd(&den[dN * 8 + h], a);
}

// GAT pass 3: agg[dst] += (alpha/den) * (v[src] + ev[e])
__global__ void edge_agg_kernel(const float* __restrict__ al, const float* __restrict__ den,
                                const float* __restrict__ v, const float* __restrict__ ev,
                                const int* __restrict__ src, const int* __restrict__ dst,
                                float* __restrict__ agg, int E)
{
    int t = blockIdx.x * 256 + threadIdx.x;
    int e = t >> 3, h = t & 7;
    if (e >= E) return;
    int sN = src[e], dN = dst[e];
    float w = al[(size_t)e * 8 + h] / (den[dN * 8 + h] + 1e-9f);
    const float* vp = v   + (size_t)sN * 256 + h * 32;
    const float* ep = ev  + (size_t)e  * 256 + h * 32;
    float*       ap = agg + (size_t)dN * 256 + h * 32;
    #pragma unroll
    for (int d = 0; d < 32; ++d) atomicAdd(&ap[d], w * (vp[d] + ep[d]));
}

// pred[tok,40] = future[tok,:] @ w[256,40] + b
__global__ void out_head_kernel(const float* __restrict__ fut, const float* __restrict__ w,
                                const float* __restrict__ b, float* __restrict__ out)
{
    int tok = blockIdx.x, o = threadIdx.x;
    if (o >= 40) return;
    float acc = b[o];
    for (int k = 0; k < 256; ++k) acc = fmaf(fut[(size_t)tok * 256 + k], w[k * 40 + o], acc);
    out[(size_t)tok * 40 + o] = acc;
}

// =====================================================================
// Host orchestration
// =====================================================================
#define GEMM(ACT, HB, HR, OF, OB, X, Wt, BIAS, RES, OUTF, OUTB, NR)                      \
    gemm256_wmma<ACT, HB, HR, OF, OB><<<dim3(((NR) + 127) / 128, 4), dim3(256), 0,       \
                                        stream>>>(X, Wt, BIAS, RES, OUTF, OUTB, NR)

extern "C" void kernel_launch(void* const* d_in, const int* in_sizes, int n_in,
                              void* d_out, int out_size, void* d_ws, size_t ws_size,
                              hipStream_t stream) {
    (void)in_sizes; (void)n_in; (void)out_size; (void)ws_size;

    const float* map_f   = (const float*)d_in[0];
    const float* agent_f = (const float*)d_in[1];
    const float* x_t     = (const float*)d_in[2];
    const float* dt      = (const float*)d_in[3];
    const float* feat_pf = (const float*)d_in[4];
    const float* feat_mf = (const float*)d_in[5];
    const float* feat_ag = (const float*)d_in[6];
    const float* feat_tmp= (const float*)d_in[7];
    const float* P       = (const float*)d_in[8];   // params: jax leaf order (sorted keys)
    const int*   ei_pf   = (const int*)d_in[9];
    const int*   ei_mf   = (const int*)d_in[10];
    const int*   ei_ag   = (const int*)d_in[11];
    const int*   ei_tmp  = (const int*)d_in[12];
    float* out = (float*)d_out;

    // ---- param offsets (jax tree flatten: dict keys sorted, lists in order) ----
    size_t po = 0;
    auto take = [&](size_t n) { size_t o = po; po += n; return o; };
    // edge_ff {b1,b2,w1,w2}
    size_t ff_b1 = take(256), ff_b2 = take(256), ff_w1 = take(5 * 256),  ff_w2 = take(65536);
    // edge_mf
    size_t mf_b1 = take(256), mf_b2 = take(256), mf_w1 = take(6 * 256),  mf_w2 = take(65536);
    // edge_pf
    size_t pf_b1 = take(256), pf_b2 = take(256), pf_w1 = take(6 * 256),  pf_w2 = take(65536);
    // gat: 3 layers x rel{ag,mf,pf,tmp} x {wek,wev,wk,wo,wq,wv}
    size_t gatOff[3][4][6];
    for (int l = 0; l < 3; ++l)
        for (int rl = 0; rl < 4; ++rl)
            for (int m = 0; m < 6; ++m) gatOff[l][rl][m] = take(65536);
    // out_head {b,w}
    size_t oh_b = take(40), oh_w = take(256 * 40);
    // seg_mlp
    size_t sg_b1 = take(256), sg_b2 = take(256), sg_w1 = take(40 * 256), sg_w2 = take(65536);
    // tc_attn {bk,bo,bq,bv,wk,wo,wq,wv}  (wq/wk/bk/bq dead: softmax over 1 key == 1)
    take(256); size_t tc_bo = take(256); take(256); size_t tc_bv = take(256);
    take(65536); size_t tc_wo = take(65536); take(65536); size_t tc_wv = take(65536);
    // time_embed
    size_t te_b1 = take(256), te_b2 = take(256), te_w1 = take(65536), te_w2 = take(65536);

    // ---- workspace layout ----
    char* base = (char*)d_ws;
    size_t off = 0;
    auto walloc = [&](size_t bytes) { size_t o = off; off = (off + bytes + 255) & ~(size_t)255; return base + o; };
    bf16*  WB      = (bf16*)walloc((size_t)80 * 65536 * 2);
    bf16*  past_bf = (bf16*)walloc((size_t)NP_N * 256 * 2);
    bf16*  map_bf  = (bf16*)walloc((size_t)M_N * 256 * 2);
    bf16*  fut_bf  = (bf16*)walloc((size_t)NF_N * 256 * 2);
    float* fut_f   = (float*)walloc((size_t)NF_N * 256 * 4);
    float* seg_f   = (float*)walloc((size_t)NF_N * 256 * 4);
    bf16*  segh_bf = (bf16*)walloc((size_t)NF_N * 256 * 2);
    bf16*  sin_bf  = (bf16*)walloc((size_t)A_N * 256 * 2);
    bf16*  th_bf   = (bf16*)walloc((size_t)A_N * 256 * 2);
    bf16*  temb_bf = (bf16*)walloc((size_t)A_N * 256 * 2);
    bf16*  vA_bf   = (bf16*)walloc((size_t)A_N * 256 * 2);
    float* oA_f    = (float*)walloc((size_t)A_N * 256 * 4);
    bf16*  a_pf    = (bf16*)walloc((size_t)E1_N * 256 * 2);
    bf16*  a_mf    = (bf16*)walloc((size_t)E1_N * 256 * 2);
    bf16*  a_ag    = (bf16*)walloc((size_t)E1_N * 256 * 2);
    bf16*  a_tmp   = (bf16*)walloc((size_t)ET_N * 256 * 2);
    bf16*  hbuf    = (bf16*)walloc((size_t)E1_N * 256 * 2);
    float* q_f     = (float*)walloc((size_t)NF_N * 256 * 4);
    float* k_f     = (float*)walloc((size_t)NP_N * 256 * 4);
    float* v_f     = (float*)walloc((size_t)NP_N * 256 * 4);
    float* ek_f    = (float*)walloc((size_t)E1_N * 256 * 4);
    float* ev_f    = (float*)walloc((size_t)E1_N * 256 * 4);
    float* s_f     = (float*)walloc((size_t)E1_N * 8 * 4);
    float* al_f    = (float*)walloc((size_t)E1_N * 8 * 4);
    unsigned* m_u  = (unsigned*)walloc((size_t)NF_N * 8 * 4);
    float* den_f   = (float*)walloc((size_t)NF_N * 8 * 4);
    float* agg_f   = (float*)walloc((size_t)NF_N * 256 * 4);
    bf16*  agg_bf  = (bf16*)walloc((size_t)NF_N * 256 * 2);

    // ---- pack all 256x256 WMMA weights into transposed bf16 ----
    int slot = 0;
    auto PACK = [&](size_t srcOff) -> bf16* {
        bf16* dst = WB + (size_t)slot * 65536;
        pack_wt_kernel<<<256, 256, 0, stream>>>(P + srcOff, dst);
        ++slot;
        return dst;
    };
    bf16* gatW[3][4][6];
    for (int l = 0; l < 3; ++l)
        for (int rl = 0; rl < 4; ++rl)
            for (int m = 0; m < 6; ++m) gatW[l][rl][m] = PACK(gatOff[l][rl][m]);
    bf16* tcWv = PACK(tc_wv); bf16* tcWo = PACK(tc_wo);
    bf16* teW1 = PACK(te_w1); bf16* teW2 = PACK(te_w2);
    bf16* pfW2 = PACK(pf_w2); bf16* mfW2 = PACK(mf_w2);
    bf16* ffW2 = PACK(ff_w2); bf16* sgW2 = PACK(sg_w2);

    // ---- bf16 mirrors of static node features ----
    f32_to_bf16_kernel<<<(NP_N * 256 + 255) / 256, 256, 0, stream>>>(agent_f, past_bf, (size_t)NP_N * 256);
    f32_to_bf16_kernel<<<(M_N * 256 + 255) / 256, 256, 0, stream>>>(map_f, map_bf, (size_t)M_N * 256);

    // ---- segment tokenization: relu(x_t@w1+b1) @ w2 + b2 ----
    mlp_in_kernel<40><<<NF_N, 256, 0, stream>>>(x_t, P + sg_w1, P + sg_b1, segh_bf);
    GEMM(0, true, false, true, false, segh_bf, sgW2, P + sg_b2, nullptr, seg_f, nullptr, NF_N);

    // ---- diffusion-time conditioning (softmax over 1 key == identity) ----
    time_sinusoid_kernel<<<A_N, 256, 0, stream>>>(dt, sin_bf);
    GEMM(2, true, false, false, true, sin_bf, teW1, P + te_b1, nullptr, nullptr, th_bf, A_N);
    GEMM(0, true, false, false, true, th_bf, teW2, P + te_b2, nullptr, nullptr, temb_bf, A_N);
    GEMM(0, true, false, false, true, temb_bf, tcWv, P + tc_bv, nullptr, nullptr, vA_bf, A_N);
    GEMM(0, true, false, true, false, vA_bf, tcWo, P + tc_bo, nullptr, oA_f, nullptr, A_N);
    seg_plus_tc_kernel<<<NF_N, 256, 0, stream>>>(seg_f, oA_f, fut_f, fut_bf);

    // ---- edge attribute MLPs -> bf16 ----
    mlp_in_kernel<6><<<E1_N, 256, 0, stream>>>(feat_pf, P + pf_w1, P + pf_b1, hbuf);
    GEMM(0, true, false, false, true, hbuf, pfW2, P + pf_b2, nullptr, nullptr, a_pf, E1_N);
    mlp_in_kernel<6><<<E1_N, 256, 0, stream>>>(feat_mf, P + mf_w1, P + mf_b1, hbuf);
    GEMM(0, true, false, false, true, hbuf, mfW2, P + mf_b2, nullptr, nullptr, a_mf, E1_N);
    mlp_in_kernel<5><<<E1_N, 256, 0, stream>>>(feat_ag, P + ff_w1, P + ff_b1, hbuf);
    GEMM(0, true, false, false, true, hbuf, ffW2, P + ff_b2, nullptr, nullptr, a_ag, E1_N);
    mlp_in_kernel<5><<<ET_N, 256, 0, stream>>>(feat_tmp, P + ff_w1, P + ff_b1, hbuf);
    GEMM(0, true, false, false, true, hbuf, ffW2, P + ff_b2, nullptr, nullptr, a_tmp, ET_N);

    // ---- 3 GAT layers x {pf, mf, ag, tmp} ----
    for (int l = 0; l < 3; ++l) {
        struct Rel { int widx; const int* ei; const bf16* xs; int ns; const bf16* aa; int E; };
        const Rel rels[4] = {
            {2, ei_pf,  past_bf, NP_N, a_pf,  E1_N},   // pf  (sorted-key index 2)
            {1, ei_mf,  map_bf,  M_N,  a_mf,  E1_N},   // mf
            {0, ei_ag,  fut_bf,  NF_N, a_ag,  E1_N},   // ag
            {3, ei_tmp, fut_bf,  NF_N, a_tmp, ET_N},   // tmp
        };
        for (int ri = 0; ri < 4; ++ri) {
            const Rel& R = rels[ri];
            bf16* Wek = gatW[l][R.widx][0];
            bf16* Wev = gatW[l][R.widx][1];
            bf16* Wk  = gatW[l][R.widx][2];
            bf16* Wo  = gatW[l][R.widx][3];
            bf16* Wq  = gatW[l][R.widx][4];
            bf16* Wv  = gatW[l][R.widx][5];
            const int* src = R.ei;
            const int* dst = R.ei + R.E;

            GEMM(0, false, false, true, false, fut_bf, Wq,  nullptr, nullptr, q_f,  nullptr, NF_N);
            GEMM(0, false, false, true, false, R.xs,   Wk,  nullptr, nullptr, k_f,  nullptr, R.ns);
            GEMM(0, false, false, true, false, R.xs,   Wv,  nullptr, nullptr, v_f,  nullptr, R.ns);
            GEMM(0, false, false, true, false, R.aa,   Wek, nullptr, nullptr, ek_f, nullptr, R.E);
            GEMM(0, false, false, true, false, R.aa,   Wev, nullptr, nullptr, ev_f, nullptr, R.E);

            fill_u32_kernel<<<(NF_N * 8 + 255) / 256, 256, 0, stream>>>(m_u, 0u, (size_t)NF_N * 8);
            fill_u32_kernel<<<(NF_N * 8 + 255) / 256, 256, 0, stream>>>((unsigned*)den_f, 0u, (size_t)NF_N * 8);
            fill_u32_kernel<<<(NF_N * 256 + 255) / 256, 256, 0, stream>>>((unsigned*)agg_f, 0u, (size_t)NF_N * 256);

            int gE = (R.E * 8 + 255) / 256;
            edge_score_kernel<<<gE, 256, 0, stream>>>(q_f, k_f, ek_f, src, dst, s_f, m_u, R.E);
            edge_exp_kernel<<<gE, 256, 0, stream>>>(s_f, dst, m_u, al_f, den_f, R.E);
            edge_agg_kernel<<<gE, 256, 0, stream>>>(al_f, den_f, v_f, ev_f, src, dst, agg_f, R.E);

            f32_to_bf16_kernel<<<(NF_N * 256 + 255) / 256, 256, 0, stream>>>(agg_f, agg_bf, (size_t)NF_N * 256);
            // future = future + agg @ wo ; refresh bf16 mirror for next relation
            GEMM(0, false, true, true, true, agg_bf, Wo, nullptr, fut_f, fut_f, fut_bf, NF_N);
        }
    }

    // ---- output head: [NF,256] @ [256,40] + b ----
    out_head_kernel<<<NF_N, 64, 0, stream>>>(fut_f, P + oh_w, P + oh_b, out);
}
